// DecoderLSTM_69999376990858
// MI455X (gfx1250) — matrix operand order repaired and checked by reading, not compiled, not run on hardware
//
#include <hip/hip_runtime.h>
#include <stdint.h>

// ---------------------------------------------------------------------------
// DecoderLSTM on gfx1250 (MI455X): persistent per-workgroup recurrence.
// B=16384, ENC=512, H=256, DE=32, T=64, gates 4H=1024.
//
//  * 256 blocks x 512 threads (16 waves). Each block owns 64 batch rows for
//    the entire T=64 recurrence; h lives in LDS (bf16), c in registers.
//  * x@W_k folded into rank-1: w_fold = W_emb@W_k, b_fold = b_emb@W_k+b_lstm.
//  * Wave w owns h-columns [16w,16w+16) and computes those columns of ALL
//    four gates -> i/f/cbar/o for one (m,j) are in the same lane: elementwise
//    LSTM update is register-only.
//  * W_r / W_enc pre-packed (setup kernel) into per-lane WMMA-B fragment
//    layout: lane L (0..15): N=n0+L, K=k0+0..15 ; lanes 16..31: same N,
//    K=k0+16..31. Consumer does two coalesced global b128 loads per fragment.
//  * A fragments (h tile, 16x32 bf16) read from LDS in the same contiguous k order as B:
//    lanes<16: K 0..15 ; lanes>=16: K 16..31 (2x ds b128).
// ---------------------------------------------------------------------------

typedef __bf16  v16bf __attribute__((ext_vector_type(16)));
typedef float   v8f   __attribute__((ext_vector_type(8)));

union FragU {
    uint4 q[2];
    v16bf v;
};

__device__ __forceinline__ unsigned short f2bf(float f) {
    uint32_t u = __float_as_uint(f);
    uint32_t r = u + 0x7FFFu + ((u >> 16) & 1u);   // round-to-nearest-even
    return (unsigned short)(r >> 16);
}
__device__ __forceinline__ float sigm(float x) {
    return 1.0f / (1.0f + __expf(-x));
}
__device__ __forceinline__ float tanh_fast(float x) {
    float e = __expf(2.0f * x);
    return (e - 1.0f) / (e + 1.0f);
}

// ---------------------------------------------------------------------------
// Setup: pack W_r (256x1024) and W_enc (512x256) into bf16 WMMA-B fragments;
// compute rank-1 folds of the DE=32 embedding path.
//   pWr  frag index: ((w*4+g)*8+kc) ; n = g*256+w*16+(L&15), k = kc*32+(L>>4)*16+e
//   pWenc frag index: (w*16+kc)     ; n = w*16+(L&15),       k = kc*32+(L>>4)*16+e
// ---------------------------------------------------------------------------
__global__ void lstm_pack_kernel(const float* __restrict__ Wr,
                                 const float* __restrict__ Wenc,
                                 const float* __restrict__ Wemb,
                                 const float* __restrict__ bemb,
                                 const float* __restrict__ Wk,
                                 const float* __restrict__ blstm,
                                 unsigned short* __restrict__ pWr,
                                 unsigned short* __restrict__ pWenc,
                                 float* __restrict__ wfold,
                                 float* __restrict__ bfold) {
    int idx    = blockIdx.x * blockDim.x + threadIdx.x;
    int stride = gridDim.x * blockDim.x;

    // W_r: 256*1024 = 262144 elements
    for (int p = idx; p < 262144; p += stride) {
        int frag = p >> 9, rem = p & 511;
        int L = rem >> 4, e = rem & 15;
        int kc = frag & 7, g = (frag >> 3) & 3, w = frag >> 5;
        int n = g * 256 + w * 16 + (L & 15);
        int k = kc * 32 + (L >> 4) * 16 + e;
        pWr[p] = f2bf(Wr[k * 1024 + n]);
    }
    // W_enc: 512*256 = 131072 elements
    for (int p = idx; p < 131072; p += stride) {
        int frag = p >> 9, rem = p & 511;
        int L = rem >> 4, e = rem & 15;
        int kc = frag & 15, w = frag >> 4;
        int n = w * 16 + (L & 15);
        int k = kc * 32 + (L >> 4) * 16 + e;
        pWenc[p] = f2bf(Wenc[k * 256 + n]);
    }
    // folds over DE=32
    for (int n = idx; n < 1024; n += stride) {
        float s = 0.0f, sb = 0.0f;
        for (int d = 0; d < 32; ++d) {
            float wk = Wk[d * 1024 + n];
            s  += Wemb[d] * wk;
            sb += bemb[d] * wk;
        }
        wfold[n] = s;
        bfold[n] = blstm[n] + sb;
    }
}

// ---------------------------------------------------------------------------
// Main kernel
// ---------------------------------------------------------------------------
__global__ __launch_bounds__(512, 1)
void lstm_main_kernel(const float* __restrict__ enc,      // [16384,512]
                      const float* __restrict__ finalx,   // [16384]
                      const float* __restrict__ Wred,     // [256]
                      const float* __restrict__ bred,     // [1]
                      const float* __restrict__ benc,     // [256]
                      const unsigned short* __restrict__ pWr,
                      const unsigned short* __restrict__ pWenc,
                      const float* __restrict__ wfold,    // [1024]
                      const float* __restrict__ bfold,    // [1024]
                      float* __restrict__ out)            // [16384,64]
{
    __shared__ __align__(16) unsigned short h_lds[64 * 256];    // 32 KB bf16 state
    __shared__ __align__(16) unsigned short enc_lds[64 * 128];  // 16 KB staging
    __shared__ float y_lds[2][64];

    const int tid  = threadIdx.x;
    const int lane = tid & 31;
    const int wv   = tid >> 5;        // 0..15
    const int half = lane >> 4;       // 0/1
    const int lcol = lane & 15;       // 0..15
    const int b0   = blockIdx.x * 64;
    const int myN  = wv * 16 + lcol;  // owned h-column

    // Per-lane constants
    const float wred_r = Wred[myN];
    const float benc_r = benc[myN];
    const float bred_r = bred[0];
    float wf_r[4], bf_r[4];
#pragma unroll
    for (int g = 0; g < 4; ++g) {
        wf_r[g] = wfold[g * 256 + myN];
        bf_r[g] = bfold[g * 256 + myN];
    }

    // ---------------- h0 = enc @ W_enc + b_enc  (K=512, 4 passes of 128) ----
    v8f hacc[4];
#pragma unroll
    for (int mt = 0; mt < 4; ++mt) { v8f z = {}; hacc[mt] = z; }

    for (int pass = 0; pass < 4; ++pass) {
        // stage [64,128] fp32 -> bf16 into LDS
        for (int i = tid; i < 64 * 128; i += 512) {
            int m = i >> 7, kk = i & 127;
            enc_lds[i] = f2bf(enc[(size_t)(b0 + m) * 512 + pass * 128 + kk]);
        }
        __syncthreads();
#pragma unroll
        for (int kcl = 0; kcl < 4; ++kcl) {
            FragU a[4];
#pragma unroll
            for (int mt = 0; mt < 4; ++mt) {
                int row  = mt * 16 + lcol;
                int koff = kcl * 32 + half * 16;
                int idx  = row * 128 + koff;
                a[mt].q[0] = *reinterpret_cast<const uint4*>(&enc_lds[idx]);
                a[mt].q[1] = *reinterpret_cast<const uint4*>(&enc_lds[idx + 8]);
            }
            int frag = wv * 16 + (pass * 4 + kcl);
            FragU bfrag;
            const uint4* gp = reinterpret_cast<const uint4*>(pWenc + frag * 512 + lane * 16);
            bfrag.q[0] = gp[0];
            bfrag.q[1] = gp[1];
#pragma unroll
            for (int mt = 0; mt < 4; ++mt) {
                hacc[mt] = __builtin_amdgcn_wmma_f32_16x16x32_bf16(
                    false, a[mt].v, false, bfrag.v, (short)0, hacc[mt], false, false);
            }
        }
        __syncthreads();
    }

    // write h0 (bf16) into LDS, init c = 0, init y0 = final_x_val
    float c[32];
#pragma unroll
    for (int mt = 0; mt < 4; ++mt) {
#pragma unroll
        for (int r = 0; r < 8; ++r) {
            int m = mt * 16 + r + half * 8;
            h_lds[m * 256 + myN] = f2bf(hacc[mt][r] + benc_r);
            c[mt * 8 + r] = 0.0f;
        }
    }
    if (tid < 64) y_lds[0][tid] = finalx[b0 + tid];
    __syncthreads();

    // ---------------- T = 64 recurrence ------------------------------------
    int rb = 0;
    for (int t = 0; t < 64; ++t) {
        const int wb = rb ^ 1;
        if (tid < 64) y_lds[wb][tid] = bred_r;   // seed y with b_red

        // gates[m, g*256+myN] = h @ W_r   (raw, fp32 accum)
        v8f acc[4][4];   // [gate][m-tile]
#pragma unroll
        for (int g = 0; g < 4; ++g)
#pragma unroll
            for (int mt = 0; mt < 4; ++mt) { v8f z = {}; acc[g][mt] = z; }

#pragma unroll
        for (int kc = 0; kc < 8; ++kc) {
            FragU a[4];
#pragma unroll
            for (int mt = 0; mt < 4; ++mt) {
                int row  = mt * 16 + lcol;
                int koff = kc * 32 + half * 16;
                int idx  = row * 256 + koff;
                a[mt].q[0] = *reinterpret_cast<const uint4*>(&h_lds[idx]);
                a[mt].q[1] = *reinterpret_cast<const uint4*>(&h_lds[idx + 8]);
            }
#pragma unroll
            for (int g = 0; g < 4; ++g) {
                int frag = (wv * 4 + g) * 8 + kc;
                FragU bfrag;
                const uint4* gp = reinterpret_cast<const uint4*>(pWr + frag * 512 + lane * 16);
                bfrag.q[0] = gp[0];
                bfrag.q[1] = gp[1];
#pragma unroll
                for (int mt = 0; mt < 4; ++mt) {
                    acc[g][mt] = __builtin_amdgcn_wmma_f32_16x16x32_bf16(
                        false, a[mt].v, false, bfrag.v, (short)0, acc[g][mt], false, false);
                }
            }
        }
        __syncthreads();   // all h_lds reads of state t done

        // rank-1 x-path + activations + c/h update, all in-lane
#pragma unroll
        for (int mt = 0; mt < 4; ++mt) {
#pragma unroll
            for (int r = 0; r < 8; ++r) {
                const int m  = mt * 16 + r + half * 8;
                const float yv = y_lds[rb][m];
                float gi = acc[0][mt][r] + yv * wf_r[0] + bf_r[0];
                float gf = acc[1][mt][r] + yv * wf_r[1] + bf_r[1];
                float gc = acc[2][mt][r] + yv * wf_r[2] + bf_r[2];
                float go = acc[3][mt][r] + yv * wf_r[3] + bf_r[3];
                float cn = sigm(gf) * c[mt * 8 + r] + sigm(gi) * tanh_fast(gc);
                c[mt * 8 + r] = cn;
                float hn = sigm(go) * tanh_fast(cn);
                h_lds[m * 256 + myN] = f2bf(hn);   // state t+1

                // y[m] += hn * W_red[myN] : reduce 16 owned columns in-wave
                float part = hn * wred_r;
                part += __shfl_xor(part, 1, 32);
                part += __shfl_xor(part, 2, 32);
                part += __shfl_xor(part, 4, 32);
                part += __shfl_xor(part, 8, 32);
                if (lcol == 0) atomicAdd(&y_lds[wb][m], part);
            }
        }
        __syncthreads();   // h(t+1), y(t) complete

        if (tid < 64) out[(size_t)(b0 + tid) * 64 + t] = y_lds[wb][tid];
        rb = wb;
    }
}

// ---------------------------------------------------------------------------
extern "C" void kernel_launch(void* const* d_in, const int* in_sizes, int n_in,
                              void* d_out, int out_size, void* d_ws, size_t ws_size,
                              hipStream_t stream) {
    const float* enc    = (const float*)d_in[0];   // [16384,512]
    const float* finalx = (const float*)d_in[1];   // [16384,1]
    const float* Wemb   = (const float*)d_in[2];   // [1,32]
    const float* bemb   = (const float*)d_in[3];   // [32]
    const float* Wenc   = (const float*)d_in[4];   // [512,256]
    const float* benc   = (const float*)d_in[5];   // [256]
    const float* Wk     = (const float*)d_in[6];   // [32,1024]
    const float* Wr     = (const float*)d_in[7];   // [256,1024]
    const float* blstm  = (const float*)d_in[8];   // [1024]
    const float* Wred   = (const float*)d_in[9];   // [256]
    const float* bred   = (const float*)d_in[10];  // [1]
    float*       out    = (float*)d_out;           // [16384,64]

    uint8_t* ws = (uint8_t*)d_ws;
    unsigned short* pWr   = (unsigned short*)(ws + 0);        // 524288 B
    unsigned short* pWenc = (unsigned short*)(ws + 524288);   // 262144 B
    float*          wfold = (float*)(ws + 786432);            //   4096 B
    float*          bfold = (float*)(ws + 790528);            //   4096 B

    lstm_pack_kernel<<<512, 256, 0, stream>>>(Wr, Wenc, Wemb, bemb, Wk, blstm,
                                              pWr, pWenc, wfold, bfold);

    lstm_main_kernel<<<256, 512, 0, stream>>>(enc, finalx, Wred, bred, benc,
                                              pWr, pWenc, wfold, bfold, out);
}
